// BitLinear2bit_73315091743457
// MI455X (gfx1250) — matrix operand order repaired and checked by reading, not compiled, not run on hardware
//
#include <hip/hip_runtime.h>
#include <hip/hip_bf16.h>

typedef _Float16 v8h  __attribute__((ext_vector_type(8)));
typedef _Float16 v16h __attribute__((ext_vector_type(16)));
typedef float    v8f  __attribute__((ext_vector_type(8)));

#define D_IN    4096
#define D_OUT   4096
#define M_TOTAL 8192
#define EPSV    1e-5f

#define BM   128
#define BN   256
#define BK   64
#define LDA  72   // padded LDS pitch in halves (72*2 = 144 B, multiple of 16 B)

// ---- CDNA5 async global->LDS copy (ASYNCcnt path), 16 B per lane ----
__device__ __forceinline__ void async_copy_b128(const void* lds_dst, const void* gsrc) {
    unsigned           ldsoff = (unsigned)(size_t)lds_dst;            // flat[31:0] == LDS offset
    unsigned long long gaddr  = (unsigned long long)(size_t)gsrc;
    asm volatile("global_load_async_to_lds_b128 %0, %1, off"
                 :: "v"(ldsoff), "v"(gaddr) : "memory");
}

// ---------------- Kernel 1: per-row absmax (row = 4096 f32) ----------------
__global__ __launch_bounds__(256)
void absmax_kernel(const float* __restrict__ x,
                   float* __restrict__ amax, float* __restrict__ rinv) {
    __shared__ float red[8];
    const int row = blockIdx.x;
    const int t = threadIdx.x;
    const float4* xr = (const float4*)(x + (size_t)row * D_IN);
    float m = 0.f;
#pragma unroll
    for (int i = 0; i < 4; ++i) {
        float4 v = xr[t + i * 256];
        m = fmaxf(m, fmaxf(fmaxf(fabsf(v.x), fabsf(v.y)),
                           fmaxf(fabsf(v.z), fabsf(v.w))));
    }
#pragma unroll
    for (int off = 16; off > 0; off >>= 1)
        m = fmaxf(m, __shfl_down(m, off, 32));
    if ((t & 31) == 0) red[t >> 5] = m;
    __syncthreads();
    if (t < 8) {
        m = red[t];
#pragma unroll
        for (int off = 4; off > 0; off >>= 1)
            m = fmaxf(m, __shfl_down(m, off, 32));
        if (t == 0) {
            m = fmaxf(m, EPSV);
            amax[row] = m;
            rinv[row] = 1.0f / m;
        }
    }
}

// ---------------- Kernel 2: x_norm -> f16 (8 elems / thread) ----------------
__global__ __launch_bounds__(256)
void xquant_kernel(const float* __restrict__ x, const float* __restrict__ rinv,
                   _Float16* __restrict__ xq) {
    const size_t idx  = (size_t)blockIdx.x * 256 + threadIdx.x;
    const size_t base = idx * 8;
    const int row = (int)(base >> 12);          // / 4096
    const float r = rinv[row];
    const float4* xp = (const float4*)(x + base);
    float4 a = xp[0], b = xp[1];
    v8h h;
    h[0] = (_Float16)(a.x * r); h[1] = (_Float16)(a.y * r);
    h[2] = (_Float16)(a.z * r); h[3] = (_Float16)(a.w * r);
    h[4] = (_Float16)(b.x * r); h[5] = (_Float16)(b.y * r);
    h[6] = (_Float16)(b.z * r); h[7] = (_Float16)(b.w * r);
    *(v8h*)(xq + base) = h;
}

// ---------------- Kernel 3: W codes -> f16 levels (code - 1.5) ----------------
__global__ __launch_bounds__(256)
void wdequant_kernel(const int* __restrict__ wq, _Float16* __restrict__ wf) {
    const size_t idx  = (size_t)blockIdx.x * 256 + threadIdx.x;
    const size_t base = idx * 8;
    const int4* p = (const int4*)(wq + base);
    int4 c0 = p[0], c1 = p[1];
    v8h h;
    h[0] = (_Float16)((float)c0.x - 1.5f); h[1] = (_Float16)((float)c0.y - 1.5f);
    h[2] = (_Float16)((float)c0.z - 1.5f); h[3] = (_Float16)((float)c0.w - 1.5f);
    h[4] = (_Float16)((float)c1.x - 1.5f); h[5] = (_Float16)((float)c1.y - 1.5f);
    h[6] = (_Float16)((float)c1.z - 1.5f); h[7] = (_Float16)((float)c1.w - 1.5f);
    *(v8h*)(wf + base) = h;
}

// -------- Kernel 4: WMMA GEMM, 128x256 block tile, 64x64 per wave, BK=64 --------
// out[m,n] = sum_k xq[m,k] * wf[n,k]   (then scaled by amax[m]*wscale, + bias[n])
__global__ __launch_bounds__(256)
void gemm_kernel(const _Float16* __restrict__ Aq, const _Float16* __restrict__ Wf,
                 const float* __restrict__ amax, const float* __restrict__ bias,
                 const float* __restrict__ wscale_p, float* __restrict__ out) {
    __shared__ _Float16 As[2][BM * LDA];   // 2 x 18 KB
    __shared__ _Float16 Bs[2][BN * LDA];   // 2 x 36 KB

    const int n0   = blockIdx.x * BN;
    const int m0   = blockIdx.y * BM;
    const int t    = threadIdx.x;
    const int wave = t >> 5;
    const int lane = t & 31;
    const int wm   = wave & 1;    // 64-row slice of the 128-row tile
    const int wn   = wave >> 1;   // 64-col slice of the 256-col tile
    const int lh   = lane >> 4;   // half-wave id
    const int r    = lane & 15;

    // Per-thread staging map: 16B chunks, 8 chunks per row
    const int srow = t >> 3;      // base row, +32 per step
    const int sc8  = t & 7;

    const v8f vzero = {0.f, 0.f, 0.f, 0.f, 0.f, 0.f, 0.f, 0.f};
    v8f acc[4][4];
#pragma unroll
    for (int mt = 0; mt < 4; ++mt)
#pragma unroll
        for (int nt = 0; nt < 4; ++nt) acc[mt][nt] = vzero;

    // Issue one K-tile: A 128 rows (4 chunks/thread) + B 256 rows (8 chunks/thread)
    // -> ASYNCcnt += 12 per wave
    auto stage = [&](int kt, int buf) {
#pragma unroll
        for (int i = 0; i < 4; ++i) {
            const int row = srow + i * 32;
            async_copy_b128(&As[buf][row * LDA + sc8 * 8],
                            Aq + (size_t)(m0 + row) * D_IN + kt + sc8 * 8);
        }
#pragma unroll
        for (int i = 0; i < 8; ++i) {
            const int row = srow + i * 32;
            async_copy_b128(&Bs[buf][row * LDA + sc8 * 8],
                            Wf + (size_t)(n0 + row) * D_IN + kt + sc8 * 8);
        }
    };

    stage(0, 0);                      // prologue: tile 0 -> buffer 0
    const int NT = D_IN / BK;         // 64 K-tiles

    for (int it = 0; it < NT; ++it) {
        const int cur = it & 1;
        if (it + 1 < NT) {
            stage((it + 1) * BK, cur ^ 1);                    // prefetch next tile
            asm volatile("s_wait_asynccnt 12" ::: "memory");  // tile 'it' landed
        } else {
            asm volatile("s_wait_asynccnt 0" ::: "memory");
        }
        __syncthreads();              // all waves' async copies for 'cur' visible

        const _Float16* Asb = &As[cur][0];
        const _Float16* Bsb = &Bs[cur][0];
#pragma unroll
        for (int kk = 0; kk < BK; kk += 32) {
            // A fragment (16x32 f16): per-lane K blocks {0-7,16-23} / {8-15,24-31}
            v16h afrag[4];
#pragma unroll
            for (int mt = 0; mt < 4; ++mt) {
                const int row = wm * 64 + mt * 16 + r;
                const v8h* p = (const v8h*)(&Asb[row * LDA + kk + lh * 8]);
                v8h lo = p[0];
                v8h hi = p[2];
                afrag[mt] = __builtin_shufflevector(lo, hi,
                    0,1,2,3,4,5,6,7,8,9,10,11,12,13,14,15);
            }
            // B fragment (32x16 f16): per-lane K blocks {0-7,16-23} / {8-15,24-31}
            v16h bfrag[4];
#pragma unroll
            for (int nt = 0; nt < 4; ++nt) {
                const int row = wn * 64 + nt * 16 + r;
                const v8h* p = (const v8h*)(&Bsb[row * LDA + kk + lh * 8]);
                v8h lo = p[0];
                v8h hi = p[2];
                bfrag[nt] = __builtin_shufflevector(lo, hi,
                    0,1,2,3,4,5,6,7,8,9,10,11,12,13,14,15);
            }
#pragma unroll
            for (int mt = 0; mt < 4; ++mt)
#pragma unroll
                for (int nt = 0; nt < 4; ++nt)
                    acc[mt][nt] = __builtin_amdgcn_wmma_f32_16x16x32_f16(
                        false, afrag[mt], false, bfrag[nt],
                        (short)0, acc[mt][nt], false, false);
        }
        __syncthreads();              // done reading 'cur' before it is refilled
    }

    // Epilogue: out = acc * (amax[m] * wscale) + bias[n]
    const float wscale = wscale_p[0];
    float ascale[4][8];
#pragma unroll
    for (int mt = 0; mt < 4; ++mt)
#pragma unroll
        for (int j = 0; j < 8; ++j)
            ascale[mt][j] = amax[m0 + wm * 64 + mt * 16 + lh * 8 + j] * wscale;
    float bv[4];
#pragma unroll
    for (int nt = 0; nt < 4; ++nt)
        bv[nt] = bias[n0 + wn * 64 + nt * 16 + r];

#pragma unroll
    for (int mt = 0; mt < 4; ++mt)
#pragma unroll
        for (int nt = 0; nt < 4; ++nt)
#pragma unroll
            for (int j = 0; j < 8; ++j) {
                const int row = m0 + wm * 64 + mt * 16 + lh * 8 + j;
                const int col = n0 + wn * 64 + nt * 16 + r;
                out[(size_t)row * D_OUT + col] = acc[mt][nt][j] * ascale[mt][j] + bv[nt];
            }
}

extern "C" void kernel_launch(void* const* d_in, const int* in_sizes, int n_in,
                              void* d_out, int out_size, void* d_ws, size_t ws_size,
                              hipStream_t stream) {
    const float* x      = (const float*)d_in[0];
    const int*   wq     = (const int*)d_in[1];
    const float* wscale = (const float*)d_in[2];
    const float* bias   = (const float*)d_in[3];
    float*       out    = (float*)d_out;

    char* ws = (char*)d_ws;
    float*    amax = (float*)ws;                                   // 32 KB
    float*    rinv = (float*)(ws + 32 * 1024);                     // 32 KB
    _Float16* xq   = (_Float16*)(ws + 64 * 1024);                  // 64 MB
    _Float16* wf   = (_Float16*)(ws + 64 * 1024 + (size_t)64 * 1024 * 1024); // 32 MB

    absmax_kernel  <<<M_TOTAL, 256, 0, stream>>>(x, amax, rinv);
    xquant_kernel  <<<(M_TOTAL * (size_t)D_IN) / (256 * 8), 256, 0, stream>>>(x, rinv, xq);
    wdequant_kernel<<<((size_t)D_OUT * D_IN) / (256 * 8), 256, 0, stream>>>(wq, wf);
    gemm_kernel    <<<dim3(D_OUT / BN, M_TOTAL / BM), 256, 0, stream>>>(
                        xq, wf, amax, bias, wscale, out);
}
